// AspectModel_11269994185097
// MI455X (gfx1250) — matrix operand order repaired and checked by reading, not compiled, not run on hardware
//
#include <hip/hip_runtime.h>
#include <hip/hip_bf16.h>

#define B_  32
#define S_  512
#define D_  768
#define H_  8
#define DK_ 96
#define L_  2
#define C_  3

typedef _Float16 v16h __attribute__((ext_vector_type(16)));
typedef _Float16 h8   __attribute__((ext_vector_type(8)));
typedef float    v8f  __attribute__((ext_vector_type(8)));

union V16H { v16h v; h8 h[2]; };

// A fragment (16x32 f16, row-major, ISA layout: lanes 0-15 M, VGPR0-3 K=kh*8+0..7,
// VGPR4-7 K=kh*8+16..23 where kh = lane>>4). Two contiguous 16B chunks per lane.
__device__ __forceinline__ v16h load_a_frag(const _Float16* tile, int lda, int lane) {
  const _Float16* p = tile + (size_t)(lane & 15) * lda + ((lane >> 4) << 3);
  V16H u;
  u.h[0] = *(const h8*)(p);
  u.h[1] = *(const h8*)(p + 16);
  return u.v;
}

// B fragment (32x16 f16) from row-major "B^T" storage Brow[N][K]:
// lane&15 selects N row; K=kh*8+0..7 and kh*8+16..23 (kh = lane>>4), matching A.
__device__ __forceinline__ v16h load_b_frag(const _Float16* tile, int ldb, int lane) {
  const _Float16* p = tile + (size_t)(lane & 15) * ldb + ((lane >> 4) << 3);
  V16H u;
  u.h[0] = *(const h8*)(p);
  u.h[1] = *(const h8*)(p + 16);
  return u.v;
}

__device__ __forceinline__ v8f wmma_f16(v16h a, v16h b, v8f c) {
  return __builtin_amdgcn_wmma_f32_16x16x32_f16(false, a, false, b, (short)0, c,
                                                false, false);
}

// ---------------- conversion kernels ----------------

__global__ void k_cvt_x(const float* __restrict__ x, _Float16* __restrict__ Xh,
                        _Float16* __restrict__ Xtr) {
  size_t idx = (size_t)blockIdx.x * blockDim.x + threadIdx.x;
  if (idx >= (size_t)B_ * S_ * D_) return;
  int d = (int)(idx % D_);
  int s = (int)((idx / D_) % S_);
  int b = (int)(idx / ((size_t)S_ * D_));
  _Float16 v = (_Float16)x[idx];
  Xh[idx] = v;
  Xtr[((size_t)b * D_ + d) * S_ + s] = v;
}

// wq/wk: [H, D, DK] -> [H*DK, D] f16 (so rows are output channels)
__global__ void k_cvt_wproj(const float* __restrict__ w, _Float16* __restrict__ Wt) {
  int idx = blockIdx.x * blockDim.x + threadIdx.x;
  if (idx >= H_ * D_ * DK_) return;
  int kk = idx % DK_;
  int d  = (idx / DK_) % D_;
  int h  = idx / (D_ * DK_);
  Wt[(size_t)(h * DK_ + kk) * D_ + d] = (_Float16)w[idx];
}

__global__ void k_cvt_f16(const float* __restrict__ src, _Float16* __restrict__ dst,
                          int n) {
  int idx = blockIdx.x * blockDim.x + threadIdx.x;
  if (idx < n) dst[idx] = (_Float16)src[idx];
}

// ---------------- Q/K projection GEMM ----------------
// C[M=16384, N=768] = Xh[M,768] @ Wt[N,768]^T ; each wave computes a 16x64 strip.
// Software-pipelined: iteration k+1 fragments are loaded before iteration k WMMAs.
__global__ void __launch_bounds__(128) k_proj(const _Float16* __restrict__ Xh,
                                              const _Float16* __restrict__ Wt,
                                              _Float16* __restrict__ Out) {
  const int lane = threadIdx.x & 31;
  const int wave = threadIdx.x >> 5;
  const int m0 = blockIdx.x * 16;
  const int n0 = blockIdx.y * 256 + wave * 64;
  const _Float16* A  = Xh + (size_t)m0 * D_;
  const _Float16* B0 = Wt + (size_t)(n0 +  0) * D_;
  const _Float16* B1 = Wt + (size_t)(n0 + 16) * D_;
  const _Float16* B2 = Wt + (size_t)(n0 + 32) * D_;
  const _Float16* B3 = Wt + (size_t)(n0 + 48) * D_;
  v8f c0 = {}, c1 = {}, c2 = {}, c3 = {};
  v16h a  = load_a_frag(A,  D_, lane);
  v16h b0 = load_b_frag(B0, D_, lane);
  v16h b1 = load_b_frag(B1, D_, lane);
  v16h b2 = load_b_frag(B2, D_, lane);
  v16h b3 = load_b_frag(B3, D_, lane);
  for (int k0 = 32; k0 < D_; k0 += 32) {
    v16h an  = load_a_frag(A + k0,  D_, lane);
    v16h b0n = load_b_frag(B0 + k0, D_, lane);
    v16h b1n = load_b_frag(B1 + k0, D_, lane);
    v16h b2n = load_b_frag(B2 + k0, D_, lane);
    v16h b3n = load_b_frag(B3 + k0, D_, lane);
    c0 = wmma_f16(a, b0, c0);
    c1 = wmma_f16(a, b1, c1);
    c2 = wmma_f16(a, b2, c2);
    c3 = wmma_f16(a, b3, c3);
    a = an; b0 = b0n; b1 = b1n; b2 = b2n; b3 = b3n;
  }
  c0 = wmma_f16(a, b0, c0);
  c1 = wmma_f16(a, b1, c1);
  c2 = wmma_f16(a, b2, c2);
  c3 = wmma_f16(a, b3, c3);
  v8f cs[4] = {c0, c1, c2, c3};
  for (int j = 0; j < 4; ++j) {
    const int n = n0 + j * 16 + (lane & 15);
    const int h = n / DK_, kk = n % DK_;
    for (int v = 0; v < 8; ++v) {
      int m = m0 + ((lane >> 4) << 3) + v;
      int bb = m >> 9, s = m & (S_ - 1);
      Out[((size_t)(bb * H_ + h) * S_ + s) * DK_ + kk] = (_Float16)cs[j][v];
    }
  }
}

// ---------------- attention -> adjacency ----------------
// grid (B, S/16); 8 waves = 8 heads; 256KB dynamic LDS: per-wave 16x512 f32 scores
__global__ void __launch_bounds__(256) k_attn_adj(const _Float16* __restrict__ Qh,
                                                  const _Float16* __restrict__ Kh,
                                                  const float* __restrict__ fmask,
                                                  _Float16* __restrict__ adjh,
                                                  float* __restrict__ rdenom) {
  extern __shared__ float smem[];  // H_ * 16 * S_ floats
  const int b = blockIdx.x;
  const int rt = blockIdx.y;
  const int wave = threadIdx.x >> 5;  // head
  const int lane = threadIdx.x & 31;
  float* sc = smem + (size_t)wave * (16 * S_);

  const _Float16* Qbase = Qh + ((size_t)(b * H_ + wave) * S_ + rt * 16) * DK_;
  const _Float16* Kbase = Kh + (size_t)(b * H_ + wave) * S_ * DK_;

  // Q row-tile fragments stay in registers across the whole column sweep
  v16h a0 = load_a_frag(Qbase + 0,  DK_, lane);
  v16h a1 = load_a_frag(Qbase + 32, DK_, lane);
  v16h a2 = load_a_frag(Qbase + 64, DK_, lane);

  const int mrow0 = (lane >> 4) << 3;
  float fr[8];
  for (int v = 0; v < 8; ++v) fr[v] = fmask[b * S_ + rt * 16 + mrow0 + v];

  const float scale = 0.10206207262f;  // 1/sqrt(96)
  // software pipeline over column tiles ((ct+1)&31 wraps the tail re-load into L2)
  v16h b0 = load_b_frag(Kbase + 0,  DK_, lane);
  v16h b1 = load_b_frag(Kbase + 32, DK_, lane);
  v16h b2 = load_b_frag(Kbase + 64, DK_, lane);
  for (int ct = 0; ct < S_ / 16; ++ct) {
    const _Float16* Bn = Kbase + (size_t)(((ct + 1) & 31) * 16) * DK_;
    v16h b0n = load_b_frag(Bn + 0,  DK_, lane);
    v16h b1n = load_b_frag(Bn + 32, DK_, lane);
    v16h b2n = load_b_frag(Bn + 64, DK_, lane);
    v8f c = {};
    c = wmma_f16(a0, b0, c);
    c = wmma_f16(a1, b1, c);
    c = wmma_f16(a2, b2, c);
    b0 = b0n; b1 = b1n; b2 = b2n;
    const int col = ct * 16 + (lane & 15);
    const float fc = fmask[b * S_ + col];
    for (int v = 0; v < 8; ++v) {
      float val = c[v] * scale;
      if (fr[v] * fc == 0.0f) val = -1e9f;
      sc[(mrow0 + v) * S_ + col] = val;
    }
  }

  // per-wave softmax over each of its 16 rows (512 cols)
  for (int i = 0; i < 16; ++i) {
    float* row = sc + i * S_;
    float m = -3.4e38f;
    for (int j = lane; j < S_; j += 32) m = fmaxf(m, row[j]);
    for (int off = 16; off > 0; off >>= 1) m = fmaxf(m, __shfl_xor(m, off, 32));
    float sum = 0.f;
    for (int j = lane; j < S_; j += 32) {
      float e = __expf(row[j] - m);
      row[j] = e;
      sum += e;
    }
    for (int off = 16; off > 0; off >>= 1) sum += __shfl_xor(sum, off, 32);
    float rinv = 1.0f / sum;
    for (int j = lane; j < S_; j += 32) row[j] *= rinv;
  }
  __syncthreads();

  // head-mean + diagonal fix + row sums; 2 rows per wave
  for (int ii = 0; ii < 2; ++ii) {
    const int i = wave * 2 + ii;
    const int r = rt * 16 + i;
    float rowsum = 0.f;
    for (int j = lane; j < S_; j += 32) {
      float acc = 0.f;
      for (int hh = 0; hh < H_; ++hh) acc += smem[(size_t)hh * (16 * S_) + i * S_ + j];
      float val = (j == r) ? 1.0f : acc * 0.125f;
      rowsum += val;
      adjh[((size_t)b * S_ + r) * S_ + j] = (_Float16)val;
    }
    for (int off = 16; off > 0; off >>= 1) rowsum += __shfl_xor(rowsum, off, 32);
    if (lane == 0) rdenom[b * S_ + r] = 1.0f / (rowsum + 1.0f);
  }
}

// ---------------- GCN GEMM 1: Ax = adj @ tokens ----------------
// A = adjh[b] [S,S]; Brow = Ttr[b] [D,S] (tokens transposed); out Axh [S,D] f16
__global__ void __launch_bounds__(128) k_gcn_ax(const _Float16* __restrict__ adjh,
                                                const _Float16* __restrict__ Ttr,
                                                _Float16* __restrict__ Axh) {
  const int lane = threadIdx.x & 31;
  const int wave = threadIdx.x >> 5;
  const int b = blockIdx.z;
  const int m0 = blockIdx.x * 16;
  const int n0 = blockIdx.y * 256 + wave * 64;
  const _Float16* A  = adjh + (size_t)b * S_ * S_ + (size_t)m0 * S_;
  const _Float16* Bb = Ttr + (size_t)b * D_ * S_;
  const _Float16* B0 = Bb + (size_t)(n0 +  0) * S_;
  const _Float16* B1 = Bb + (size_t)(n0 + 16) * S_;
  const _Float16* B2 = Bb + (size_t)(n0 + 32) * S_;
  const _Float16* B3 = Bb + (size_t)(n0 + 48) * S_;
  v8f c0 = {}, c1 = {}, c2 = {}, c3 = {};
  v16h a  = load_a_frag(A,  S_, lane);
  v16h b0 = load_b_frag(B0, S_, lane);
  v16h b1 = load_b_frag(B1, S_, lane);
  v16h b2 = load_b_frag(B2, S_, lane);
  v16h b3 = load_b_frag(B3, S_, lane);
  for (int k0 = 32; k0 < S_; k0 += 32) {
    v16h an  = load_a_frag(A + k0,  S_, lane);
    v16h b0n = load_b_frag(B0 + k0, S_, lane);
    v16h b1n = load_b_frag(B1 + k0, S_, lane);
    v16h b2n = load_b_frag(B2 + k0, S_, lane);
    v16h b3n = load_b_frag(B3 + k0, S_, lane);
    c0 = wmma_f16(a, b0, c0);
    c1 = wmma_f16(a, b1, c1);
    c2 = wmma_f16(a, b2, c2);
    c3 = wmma_f16(a, b3, c3);
    a = an; b0 = b0n; b1 = b1n; b2 = b2n; b3 = b3n;
  }
  c0 = wmma_f16(a, b0, c0);
  c1 = wmma_f16(a, b1, c1);
  c2 = wmma_f16(a, b2, c2);
  c3 = wmma_f16(a, b3, c3);
  v8f cs[4] = {c0, c1, c2, c3};
  _Float16* out = Axh + (size_t)b * S_ * D_;
  for (int j = 0; j < 4; ++j) {
    const int n = n0 + j * 16 + (lane & 15);
    for (int v = 0; v < 8; ++v) {
      int m = m0 + ((lane >> 4) << 3) + v;
      out[(size_t)m * D_ + n] = (_Float16)cs[j][v];
    }
  }
}

// ---------------- GCN GEMM 2: tokens' = relu((Ax @ G^T + b) / denom) ----------------
// A = Axh[b] [S,D]; Brow = G [D,D] (gcn_w is already [e,d]); write transposed [D,S]
__global__ void __launch_bounds__(128) k_gcn_axw(const _Float16* __restrict__ Axh,
                                                 const _Float16* __restrict__ G,
                                                 const float* __restrict__ gb,
                                                 const float* __restrict__ rdenom,
                                                 _Float16* __restrict__ Tout) {
  const int lane = threadIdx.x & 31;
  const int wave = threadIdx.x >> 5;
  const int b = blockIdx.z;
  const int m0 = blockIdx.x * 16;
  const int n0 = blockIdx.y * 256 + wave * 64;
  const _Float16* A  = Axh + (size_t)b * S_ * D_ + (size_t)m0 * D_;
  const _Float16* B0 = G + (size_t)(n0 +  0) * D_;
  const _Float16* B1 = G + (size_t)(n0 + 16) * D_;
  const _Float16* B2 = G + (size_t)(n0 + 32) * D_;
  const _Float16* B3 = G + (size_t)(n0 + 48) * D_;
  v8f c0 = {}, c1 = {}, c2 = {}, c3 = {};
  v16h a  = load_a_frag(A,  D_, lane);
  v16h b0 = load_b_frag(B0, D_, lane);
  v16h b1 = load_b_frag(B1, D_, lane);
  v16h b2 = load_b_frag(B2, D_, lane);
  v16h b3 = load_b_frag(B3, D_, lane);
  for (int k0 = 32; k0 < D_; k0 += 32) {
    v16h an  = load_a_frag(A + k0,  D_, lane);
    v16h b0n = load_b_frag(B0 + k0, D_, lane);
    v16h b1n = load_b_frag(B1 + k0, D_, lane);
    v16h b2n = load_b_frag(B2 + k0, D_, lane);
    v16h b3n = load_b_frag(B3 + k0, D_, lane);
    c0 = wmma_f16(a, b0, c0);
    c1 = wmma_f16(a, b1, c1);
    c2 = wmma_f16(a, b2, c2);
    c3 = wmma_f16(a, b3, c3);
    a = an; b0 = b0n; b1 = b1n; b2 = b2n; b3 = b3n;
  }
  c0 = wmma_f16(a, b0, c0);
  c1 = wmma_f16(a, b1, c1);
  c2 = wmma_f16(a, b2, c2);
  c3 = wmma_f16(a, b3, c3);
  v8f cs[4] = {c0, c1, c2, c3};
  float rd[8];
  for (int v = 0; v < 8; ++v)
    rd[v] = rdenom[b * S_ + m0 + ((lane >> 4) << 3) + v];
  for (int j = 0; j < 4; ++j) {
    const int n = n0 + j * 16 + (lane & 15);
    const float bias = gb[n];
    _Float16* out = Tout + (size_t)b * D_ * S_ + (size_t)n * S_;
    for (int v = 0; v < 8; ++v) {
      int m = m0 + ((lane >> 4) << 3) + v;
      float val = (cs[j][v] + bias) * rd[v];
      out[m] = (_Float16)fmaxf(val, 0.0f);
    }
  }
}

// ---------------- aspect max-pool: one wave per (b,d) ----------------
__global__ void __launch_bounds__(256) k_pool(const _Float16* __restrict__ Ttr,
                                              const int* __restrict__ amask,
                                              float* __restrict__ preds) {
  const int lane = threadIdx.x & 31;
  const int row = blockIdx.x * 8 + (threadIdx.x >> 5);  // row in [0, B*D)
  const int b = row / D_;
  const _Float16* p = Ttr + (size_t)row * S_;  // Ttr is [B, D, S]
  float m = -3.4e38f;
  for (int s = lane; s < S_; s += 32) {
    float t = (amask[b * S_ + s] == 1) ? (float)p[s] : -10000.0f;
    m = fmaxf(m, t);
  }
  for (int off = 16; off > 0; off >>= 1) m = fmaxf(m, __shfl_xor(m, off, 32));
  if (lane == 0) preds[row] = m;
}

// ---------------- final tiny FFN (fp32, negligible cost) ----------------
__global__ void __launch_bounds__(256) k_ffn(const float* __restrict__ preds,
                                             const float* __restrict__ w1,
                                             const float* __restrict__ b1,
                                             const float* __restrict__ w2,
                                             const float* __restrict__ b2,
                                             float* __restrict__ out) {
  __shared__ float sx[D_];
  __shared__ float sh[D_];
  __shared__ float red[256];
  const int b = blockIdx.x, tid = threadIdx.x;
  for (int d = tid; d < D_; d += 256) sx[d] = preds[b * D_ + d];
  __syncthreads();
  for (int j = tid; j < D_; j += 256) {
    float acc = b1[j];
    const float* wr = w1 + (size_t)j * D_;
    for (int d = 0; d < D_; ++d) acc += sx[d] * wr[d];
    sh[j] = fmaxf(acc, 0.0f);
  }
  __syncthreads();
  for (int c = 0; c < C_; ++c) {
    float acc = 0.f;
    const float* wr = w2 + (size_t)c * D_;
    for (int j = tid; j < D_; j += 256) acc += sh[j] * wr[j];
    red[tid] = acc;
    __syncthreads();
    for (int s = 128; s > 0; s >>= 1) {
      if (tid < s) red[tid] += red[tid + s];
      __syncthreads();
    }
    if (tid == 0) out[b * C_ + c] = red[0] + b2[c];
    __syncthreads();
  }
}

extern "C" void kernel_launch(void* const* d_in, const int* in_sizes, int n_in,
                              void* d_out, int out_size, void* d_ws, size_t ws_size,
                              hipStream_t stream) {
  (void)in_sizes; (void)n_in; (void)out_size; (void)ws_size;
  const float* x      = (const float*)d_in[0];
  const float* fmask  = (const float*)d_in[1];
  const int*   amask  = (const int*)d_in[2];
  const float* wq     = (const float*)d_in[3];
  const float* wk     = (const float*)d_in[4];
  const float* gcn_w  = (const float*)d_in[5];
  const float* gcn_b  = (const float*)d_in[6];
  const float* ffn_w1 = (const float*)d_in[7];
  const float* ffn_b1 = (const float*)d_in[8];
  const float* ffn_w2 = (const float*)d_in[9];
  const float* ffn_b2 = (const float*)d_in[10];

  float* out   = (float*)d_out;     // [B*C] ++ [B*D]
  float* preds = out + B_ * C_;

  char* w = (char*)d_ws;
  auto alloc = [&](size_t bytes) -> char* {
    char* p = w;
    w += (bytes + 255) & ~(size_t)255;
    return p;
  };
  _Float16* Xh   = (_Float16*)alloc((size_t)B_ * S_ * D_ * 2);
  _Float16* Xtr  = (_Float16*)alloc((size_t)B_ * D_ * S_ * 2);
  _Float16* Wqh  = (_Float16*)alloc((size_t)H_ * DK_ * D_ * 2);
  _Float16* Wkh  = (_Float16*)alloc((size_t)H_ * DK_ * D_ * 2);
  _Float16* Gh   = (_Float16*)alloc((size_t)L_ * D_ * D_ * 2);
  _Float16* Qh   = (_Float16*)alloc((size_t)B_ * H_ * S_ * DK_ * 2);
  _Float16* Kh   = (_Float16*)alloc((size_t)B_ * H_ * S_ * DK_ * 2);
  _Float16* adjh = (_Float16*)alloc((size_t)B_ * S_ * S_ * 2);
  _Float16* Axh  = (_Float16*)alloc((size_t)B_ * S_ * D_ * 2);
  _Float16* Ttr1 = (_Float16*)alloc((size_t)B_ * D_ * S_ * 2);
  _Float16* Ttr2 = (_Float16*)alloc((size_t)B_ * D_ * S_ * 2);
  float*    rden = (float*)alloc((size_t)B_ * S_ * 4);

  {
    int n = B_ * S_ * D_;
    k_cvt_x<<<(n + 255) / 256, 256, 0, stream>>>(x, Xh, Xtr);
  }
  {
    int n = H_ * D_ * DK_;
    k_cvt_wproj<<<(n + 255) / 256, 256, 0, stream>>>(wq, Wqh);
    k_cvt_wproj<<<(n + 255) / 256, 256, 0, stream>>>(wk, Wkh);
  }
  {
    int n = L_ * D_ * D_;
    k_cvt_f16<<<(n + 255) / 256, 256, 0, stream>>>(gcn_w, Gh, n);
  }

  dim3 gp(B_ * S_ / 16, (H_ * DK_) / 256);   // (1024, 3); wave = 16x64 strip
  k_proj<<<gp, 128, 0, stream>>>(Xh, Wqh, Qh);
  k_proj<<<gp, 128, 0, stream>>>(Xh, Wkh, Kh);

  dim3 ga(B_, S_ / 16);
  size_t shbytes = (size_t)H_ * 16 * S_ * sizeof(float);  // 256 KB (WGP has 320 KB)
  k_attn_adj<<<ga, 256, shbytes, stream>>>(Qh, Kh, fmask, adjh, rden);

  dim3 gg(S_ / 16, D_ / 256, B_);            // (32, 3, 32)
  // layer 0
  k_gcn_ax<<<gg, 128, 0, stream>>>(adjh, Xtr, Axh);
  k_gcn_axw<<<gg, 128, 0, stream>>>(Axh, Gh, gcn_b, rden, Ttr1);
  // layer 1
  k_gcn_ax<<<gg, 128, 0, stream>>>(adjh, Ttr1, Axh);
  k_gcn_axw<<<gg, 128, 0, stream>>>(Axh, Gh + (size_t)D_ * D_, gcn_b + D_, rden, Ttr2);

  k_pool<<<(B_ * D_) / 8, 256, 0, stream>>>(Ttr2, amask, preds);
  k_ffn<<<B_, 256, 0, stream>>>(preds, ffn_w1, ffn_b1, ffn_w2, ffn_b2, out);
}
